// ProjectedAdaptiveLogSoftmax_66236985639681
// MI455X (gfx1250) — compile-verified, matrix operand order checked by reading, not run on hardware
//
#include <hip/hip_runtime.h>
#include <hip/hip_bf16.h>
#include <float.h>

// ---------------------------------------------------------------------------
// ProjectedAdaptiveLogSoftmax NLL for MI455X (gfx1250), wave32 + WMMA bf16.
// Split-fp32 (bf16 hi/lo, 3 WMMAs) for ~fp32 accuracy at bf16-WMMA rate.
// W is pre-split to bf16 hi/lo once (if workspace allows) so the streaming
// LSE loop is pure loads + WMMA; 64 tokens/block amortize B-tile traffic.
// ---------------------------------------------------------------------------

typedef __attribute__((ext_vector_type(16))) __bf16 v16bf;
typedef __attribute__((ext_vector_type(8)))  __bf16 v8bf;
typedef __attribute__((ext_vector_type(8)))  float  v8f;
typedef __attribute__((ext_vector_type(4)))  float  v4f;

union V16BF { v16bf v; v8bf h[2]; };

#define NTOK 1024
#define DHID 512
#define CUT1 20000
#define CUT2 40000
#define CUT3 200000

__device__ __forceinline__ void split_bf16(float x, __bf16 &hi, __bf16 &lo) {
    hi = (__bf16)x;
    lo = (__bf16)(x - (float)hi);
}

// online log-sum-exp merge: state (m,s) <- merge (mo,so); s==0 means empty
__device__ __forceinline__ void smerge(float &m, float &s, float mo, float so) {
    if (so > 0.0f) {
        if (s > 0.0f) {
            float M = fmaxf(m, mo);
            s = s * __expf(m - M) + so * __expf(mo - M);
            m = M;
        } else { m = mo; s = so; }
    }
}

__device__ __forceinline__ v8f wmma_bf16(const V16BF &a, const V16BF &b, v8f c) {
    return __builtin_amdgcn_wmma_f32_16x16x32_bf16(false, a.v, false, b.v,
                                                   (short)0, c, false, false);
}

// ---------------------------------------------------------------------------
// Stage 1: P_c = hidden @ proj_c  (K = 512), split into bf16 hi/lo + fp32,
// stored row-major [token][kpad]. One wave per block; proj column tile is
// staged (zero-padded) in LDS with vectorized loads.
// blockIdx.x = token tile, blockIdx.y = 16-col tile.
// ---------------------------------------------------------------------------
__global__ __launch_bounds__(32)
void proj_split_kernel(const float* __restrict__ hidden,
                       const float* __restrict__ proj,   // [512][d]
                       __bf16* __restrict__ phi,
                       __bf16* __restrict__ plo,
                       float*  __restrict__ pf32,        // [1024][kpad]
                       int d, int kpad)
{
    __shared__ float ptile[DHID * 16];                   // [k][16 cols], 32 KB

    const int lane = threadIdx.x & 31;
    const int tt   = blockIdx.x;
    const int ct   = blockIdx.y;
    const int n    = lane & 15;
    const int hl   = lane >> 4;
    const int col  = ct * 16 + n;               // output column (P's K index)
    const int arow = tt * 16 + n;               // A row (token, M = lane%16)

    // stage proj[:, ct*16 .. ct*16+15] into LDS (zero-padded beyond d)
    for (int i4 = lane; i4 < DHID * 4; i4 += 32) {
        const int k = i4 >> 2, seg = i4 & 3;
        const int cbase = ct * 16 + seg * 4;
        v4f v;
        if (cbase + 3 < d) {
            v = *(const v4f*)&proj[k * d + cbase];
        } else {
            #pragma unroll
            for (int j = 0; j < 4; ++j)
                v[j] = (cbase + j < d) ? proj[k * d + cbase + j] : 0.0f;
        }
        *(v4f*)&ptile[k * 16 + seg * 4] = v;
    }
    __syncthreads();

    v8f acc = {};
    #pragma unroll 4
    for (int kb = 0; kb < DHID / 32; ++kb) {
        const int kc = kb * 32 + hl * 8;
        // A: hidden[arow][kc..kc+7] and [kc+16..kc+23]  (16-bit A layout)
        v4f a0 = *(const v4f*)&hidden[arow * DHID + kc];
        v4f a1 = *(const v4f*)&hidden[arow * DHID + kc + 4];
        v4f a2 = *(const v4f*)&hidden[arow * DHID + kc + 16];
        v4f a3 = *(const v4f*)&hidden[arow * DHID + kc + 20];
        float af[16];
        #pragma unroll
        for (int j = 0; j < 4; ++j) {
            af[j] = a0[j]; af[4 + j] = a1[j]; af[8 + j] = a2[j]; af[12 + j] = a3[j];
        }
        V16BF ahi, alo, bhi, blo;
        #pragma unroll
        for (int e = 0; e < 16; ++e) {
            __bf16 h, l; split_bf16(af[e], h, l);
            ahi.v[e] = h; alo.v[e] = l;
        }
        // B: proj tile from LDS (already zero-padded)
        #pragma unroll
        for (int e = 0; e < 16; ++e) {
            int k = kc + (e < 8 ? e : e + 8);
            float f = ptile[k * 16 + n];
            __bf16 h, l; split_bf16(f, h, l);
            bhi.v[e] = h; blo.v[e] = l;
        }
        acc = wmma_bf16(ahi, bhi, acc);
        acc = wmma_bf16(ahi, blo, acc);
        acc = wmma_bf16(alo, bhi, acc);
    }
    // D layout: lane covers N=col, rows M = r + 8*hl
    #pragma unroll
    for (int r = 0; r < 8; ++r) {
        int tok = tt * 16 + r + 8 * hl;
        float x = acc[r];
        __bf16 h, l; split_bf16(x, h, l);
        phi [(size_t)tok * kpad + col] = h;
        plo [(size_t)tok * kpad + col] = l;
        pf32[(size_t)tok * kpad + col] = x;
    }
}

// ---------------------------------------------------------------------------
// Stage 1.5: pre-split W (fp32 [vocab][Kv]) into bf16 hi/lo [vocab][kpad],
// zero-padded to kpad. Each thread handles 8 consecutive K elements.
// ---------------------------------------------------------------------------
__global__ __launch_bounds__(256)
void wsplit_kernel(const float* __restrict__ W,
                   __bf16* __restrict__ whi, __bf16* __restrict__ wlo,
                   int vocab, int Kv, int kpad)
{
    const int idx = blockIdx.x * 256 + threadIdx.x;
    const int nc8 = kpad >> 3;
    const int row = idx / nc8;
    const int c8  = (idx - row * nc8) << 3;
    if (row >= vocab) return;
    const float* wr = &W[(size_t)row * Kv];
    float f[8];
    if (c8 + 7 < Kv) {
        v4f a = *(const v4f*)&wr[c8];
        v4f b = *(const v4f*)&wr[c8 + 4];
        #pragma unroll
        for (int j = 0; j < 4; ++j) { f[j] = a[j]; f[4 + j] = b[j]; }
    } else {
        #pragma unroll
        for (int j = 0; j < 8; ++j) f[j] = (c8 + j < Kv) ? wr[c8 + j] : 0.0f;
    }
    v8bf h, l;
    #pragma unroll
    for (int j = 0; j < 8; ++j) { __bf16 hh, ll; split_bf16(f[j], hh, ll); h[j] = hh; l[j] = ll; }
    *(v8bf*)&whi[(size_t)row * kpad + c8] = h;
    *(v8bf*)&wlo[(size_t)row * kpad + c8] = l;
}

// ---------------------------------------------------------------------------
// Stage 2: streaming logits + online softmax (per cluster).
// block = 8 waves; owns 64 tokens (4 A-subtiles in LDS) so each B tile feeds
// 12 WMMAs. blockIdx.x = token group (16), blockIdx.y = stripe.
// Emits one (m, s) partial per (stripe, token).
// PRECONV selects the pre-split-W fast path at compile time.
// ---------------------------------------------------------------------------
template<bool PRECONV>
__global__ __launch_bounds__(256)
void lse_kernel(const __bf16* __restrict__ phi,
                const __bf16* __restrict__ plo,
                const float*  __restrict__ W,      // [vocab][Kv]  (fallback)
                const __bf16* __restrict__ whi,    // [vocab][kpad] (pre-split)
                const __bf16* __restrict__ wlo,
                const float*  __restrict__ bias,   // [vocab]
                int vocab, int Kv, int kpad, int nstripes,
                float* __restrict__ pm, float* __restrict__ ps)
{
    extern __shared__ __align__(32) char smem[];
    __bf16* Ahi  = (__bf16*)smem;                   // [64][kpad]
    __bf16* Alo  = Ahi + 64 * kpad;
    float*  mArr = (float*)(Alo + 64 * kpad);       // [8][64]
    float*  sArr = mArr + 512;

    const int tid    = threadIdx.x;
    const int lane   = tid & 31;
    const int wave   = tid >> 5;
    const int g      = blockIdx.x;                  // token group (64 tokens)
    const int stripe = blockIdx.y;

    // stage the 64-token P tile (bf16 hi/lo) into LDS
    {
        const uint4* srcH = (const uint4*)(phi + (size_t)g * 64 * kpad);
        const uint4* srcL = (const uint4*)(plo + (size_t)g * 64 * kpad);
        uint4* dH = (uint4*)Ahi; uint4* dL = (uint4*)Alo;
        const int n16 = (64 * kpad) / 8;            // uint4 = 8 bf16
        for (int i = tid; i < n16; i += 256) { dH[i] = srcH[i]; dL[i] = srcL[i]; }
    }
    __syncthreads();

    const int n  = lane & 15;
    const int hl = lane >> 4;

    float m[32], s[32];
    #pragma unroll
    for (int i = 0; i < 32; ++i) { m[i] = -FLT_MAX; s[i] = 0.0f; }

    const int nvt    = (vocab + 15) >> 4;
    const int nkb    = kpad >> 5;
    const int stride = nstripes * 8;

    for (int tile = stripe * 8 + wave; tile < nvt; tile += stride) {
        const int  row   = tile * 16 + n;           // relative vocab index
        const bool rowok = row < vocab;
        const int  rowc  = rowok ? row : vocab - 1; // clamped (branchless loads)
        {   // prefetch next tile's row
            int nr = (tile + stride) * 16 + n;
            if (nr < vocab) {
                if (PRECONV) __builtin_prefetch(&whi[(size_t)nr * kpad], 0, 1);
                else         __builtin_prefetch(&W  [(size_t)nr * Kv],  0, 1);
            }
        }
        v8f acc[4];
        #pragma unroll
        for (int sub = 0; sub < 4; ++sub) acc[sub] = (v8f){};

        for (int kb = 0; kb < nkb; ++kb) {
            const int kc = kb * 32 + hl * 8;
            V16BF bhi, blo;
            if (PRECONV) {
                const __bf16* wh = &whi[(size_t)rowc * kpad + kc];
                const __bf16* wl = &wlo[(size_t)rowc * kpad + kc];
                bhi.h[0] = *(const v8bf*)wh;  bhi.h[1] = *(const v8bf*)(wh + 16);
                blo.h[0] = *(const v8bf*)wl;  blo.h[1] = *(const v8bf*)(wl + 16);
            } else {
                const float* wr = &W[(size_t)rowc * Kv];
                float f[16];
                if (kc + 23 < Kv) {
                    v4f b0 = *(const v4f*)&wr[kc];
                    v4f b1 = *(const v4f*)&wr[kc + 4];
                    v4f b2 = *(const v4f*)&wr[kc + 16];
                    v4f b3 = *(const v4f*)&wr[kc + 20];
                    #pragma unroll
                    for (int j = 0; j < 4; ++j) {
                        f[j] = b0[j]; f[4 + j] = b1[j]; f[8 + j] = b2[j]; f[12 + j] = b3[j];
                    }
                } else {
                    #pragma unroll
                    for (int e = 0; e < 16; ++e) {
                        int k = kc + (e < 8 ? e : e + 8);
                        f[e] = (k < Kv) ? wr[k] : 0.0f;
                    }
                }
                #pragma unroll
                for (int e = 0; e < 16; ++e) {
                    __bf16 h, l; split_bf16(f[e], h, l);
                    bhi.v[e] = h; blo.v[e] = l;
                }
            }
            #pragma unroll
            for (int sub = 0; sub < 4; ++sub) {
                V16BF ahi, alo;
                const __bf16* ah = &Ahi[(sub * 16 + n) * kpad + kc];
                const __bf16* al = &Alo[(sub * 16 + n) * kpad + kc];
                ahi.h[0] = *(const v8bf*)ah;  ahi.h[1] = *(const v8bf*)(ah + 16);
                alo.h[0] = *(const v8bf*)al;  alo.h[1] = *(const v8bf*)(al + 16);
                acc[sub] = wmma_bf16(ahi, bhi, acc[sub]);
                acc[sub] = wmma_bf16(ahi, blo, acc[sub]);
                acc[sub] = wmma_bf16(alo, bhi, acc[sub]);
            }
        }
        if (rowok) {
            float bv = bias[row];
            #pragma unroll
            for (int sub = 0; sub < 4; ++sub) {
                #pragma unroll
                for (int r = 0; r < 8; ++r) {
                    float x  = acc[sub][r] + bv;
                    int   i  = sub * 8 + r;
                    float mn = fmaxf(m[i], x);               // branchless online
                    s[i] = s[i] * __expf(m[i] - mn) + __expf(x - mn);
                    m[i] = mn;
                }
            }
        }
    }

    // butterfly reduce across the 16 lanes sharing each M half
    #pragma unroll
    for (int off = 1; off < 16; off <<= 1) {
        #pragma unroll
        for (int i = 0; i < 32; ++i) {
            float mo = __shfl_xor(m[i], off, 32);
            float so = __shfl_xor(s[i], off, 32);
            smerge(m[i], s[i], mo, so);
        }
    }
    if (n == 0) {   // lanes 0 (M 0-7) and 16 (M 8-15) of each wave
        #pragma unroll
        for (int sub = 0; sub < 4; ++sub)
            #pragma unroll
            for (int r = 0; r < 8; ++r) {
                int idx = wave * 64 + sub * 16 + hl * 8 + r;
                mArr[idx] = m[sub * 8 + r];
                sArr[idx] = s[sub * 8 + r];
            }
    }
    __syncthreads();

    if (tid < 64) {     // merge the 8 waves, one partial per token
        float M = -FLT_MAX, S = 0.0f;
        for (int w = 0; w < 8; ++w)
            smerge(M, S, mArr[w * 64 + tid], sArr[w * 64 + tid]);
        size_t o = (size_t)stripe * NTOK + g * 64 + tid;
        pm[o] = M; ps[o] = S;
    }
}

// ---------------------------------------------------------------------------
// Stage 3: merge stripe partials, add the 3 cluster-head logits, compute the
// target logit as an exact fp32 dot, emit nll.
// ---------------------------------------------------------------------------
__global__ __launch_bounds__(256)
void finalize_kernel(const float* __restrict__ pm, const float* __restrict__ ps,
                     const float* __restrict__ pf,        // split-P fp32 bank
                     const float* __restrict__ W0, const float* __restrict__ W1,
                     const float* __restrict__ W2, const float* __restrict__ W3,
                     const float* __restrict__ b0, const float* __restrict__ b1,
                     const float* __restrict__ b2, const float* __restrict__ b3,
                     const float* __restrict__ cw,        // [3][512]
                     const float* __restrict__ cb,        // [3]
                     const int*   __restrict__ target,
                     float* __restrict__ out)
{
    int tok = blockIdx.x * blockDim.x + threadIdx.x;
    if (tok >= NTOK) return;
    const int SOFF[5] = {0, 32, 48, 88, 104};
    const int dsz[4]  = {512, 128, 32, 8};
    const int kpad[4] = {512, 128, 32, 32};
    const int cutL[4] = {0, CUT1, CUT2, CUT3};
    const size_t pko[4] = {0, (size_t)NTOK * 512, (size_t)NTOK * 640, (size_t)NTOK * 672};

    float lse[4], clog[3] = {0.f, 0.f, 0.f};
    for (int c = 0; c < 4; ++c) {
        float M = -FLT_MAX, S = 0.0f;
        for (int st = SOFF[c]; st < SOFF[c + 1]; ++st) {
            size_t o = (size_t)st * NTOK + tok;
            smerge(M, S, pm[o], ps[o]);
        }
        if (c == 0) {   // fold the three cluster columns into the head LSE
            const float* p0 = pf + (size_t)tok * 512;
            for (int j = 0; j < 3; ++j) {
                float acc = 0.0f;
                for (int k = 0; k < DHID; ++k)
                    acc = fmaf(p0[k], cw[j * DHID + k], acc);
                clog[j] = acc + cb[j];
                smerge(M, S, clog[j], 1.0f);
            }
        }
        lse[c] = M + __logf(S);
    }
    int t   = target[tok];
    int cid = (t >= CUT1) + (t >= CUT2) + (t >= CUT3);
    int rel = t - cutL[cid];
    const float* Wc = (cid == 0) ? W0 : (cid == 1) ? W1 : (cid == 2) ? W2 : W3;
    const float* bc = (cid == 0) ? b0 : (cid == 1) ? b1 : (cid == 2) ? b2 : b3;
    int d = dsz[cid];
    const float* pv = pf + pko[cid] + (size_t)tok * kpad[cid];
    float acc = 0.0f;
    for (int k = 0; k < d; ++k)
        acc = fmaf(pv[k], Wc[(size_t)rel * d + k], acc);
    float tlog = acc + bc[rel];

    float nll;
    if (cid == 0) nll = lse[0] - tlog;
    else          nll = (lse[0] - clog[3 - cid]) + (lse[cid] - tlog);
    out[tok] = nll;
}

// ---------------------------------------------------------------------------
extern "C" void kernel_launch(void* const* d_in, const int* in_sizes, int n_in,
                              void* d_out, int out_size, void* d_ws, size_t ws_size,
                              hipStream_t stream)
{
    (void)in_sizes; (void)n_in; (void)out_size;

    const float* hidden = (const float*)d_in[0];
    const int*   target = (const int*)  d_in[1];
    const float* W[4] = {(const float*)d_in[2],  (const float*)d_in[5],
                         (const float*)d_in[8],  (const float*)d_in[11]};
    const float* B[4] = {(const float*)d_in[3],  (const float*)d_in[6],
                         (const float*)d_in[9],  (const float*)d_in[12]};
    const float* P[4] = {(const float*)d_in[4],  (const float*)d_in[7],
                         (const float*)d_in[10], (const float*)d_in[13]};
    const float* cw = (const float*)d_in[14];
    const float* cb = (const float*)d_in[15];
    float* out = (float*)d_out;

    const int dsz[4]   = {512, 128, 32, 8};
    const int kpad[4]  = {512, 128, 32, 32};
    const int vocab[4] = {20000, 20000, 160000, 67735};
    const int S[4]     = {32, 16, 40, 16};
    const int soff[4]  = {0, 32, 48, 88};
    const int NSTR     = 104;

    // workspace carve-up (256B aligned)
    char* wsb = (char*)d_ws;
    size_t off = 0;
    auto take = [&](size_t bytes) {
        size_t o = off; off = (off + bytes + 255) & ~(size_t)255; return o;
    };
    float*  pf  = (float*) (wsb + take((size_t)NTOK * 704 * 4));
    __bf16* phi = (__bf16*)(wsb + take((size_t)NTOK * 704 * 2));
    __bf16* plo = (__bf16*)(wsb + take((size_t)NTOK * 704 * 2));
    float*  pm  = (float*) (wsb + take((size_t)NSTR * NTOK * 4));
    float*  psv = (float*) (wsb + take((size_t)NSTR * NTOK * 4));
    // pre-split W bank (optional, big)
    const size_t wko[4]  = {0, 10240000, 12800000, 17920000};   // element offsets
    const size_t welems  = 20087520;                            // total elements
    __bf16* whi = (__bf16*)(wsb + take(welems * 2));
    __bf16* wlo = (__bf16*)(wsb + take(welems * 2));
    const bool preconv = (off <= ws_size);

    const size_t pko[4] = {0, (size_t)NTOK * 512, (size_t)NTOK * 640, (size_t)NTOK * 672};

    // stage 1: projections (split bf16 + fp32)
    for (int c = 0; c < 4; ++c) {
        dim3 g(NTOK / 16, kpad[c] / 16);
        proj_split_kernel<<<g, 32, 0, stream>>>(hidden, P[c], phi + pko[c], plo + pko[c],
                                                pf + pko[c], dsz[c], kpad[c]);
    }
    // stage 1.5: pre-split W into bf16 hi/lo (if workspace allows)
    if (preconv) {
        for (int c = 0; c < 4; ++c) {
            int nthr = vocab[c] * (kpad[c] >> 3);
            wsplit_kernel<<<(nthr + 255) / 256, 256, 0, stream>>>(
                W[c], whi + wko[c], wlo + wko[c], vocab[c], dsz[c], kpad[c]);
        }
    }
    // stage 2: per-cluster streaming logits + online softmax
    for (int c = 0; c < 4; ++c) {
        dim3 g(NTOK / 64, S[c]);
        size_t sm = (size_t)64 * kpad[c] * 2 * 2 + 512 * 4 * 2;
        if (preconv) {
            lse_kernel<true><<<g, 256, sm, stream>>>(
                phi + pko[c], plo + pko[c], W[c], whi + wko[c], wlo + wko[c], B[c],
                vocab[c], dsz[c], kpad[c], S[c],
                pm + (size_t)soff[c] * NTOK, psv + (size_t)soff[c] * NTOK);
        } else {
            lse_kernel<false><<<g, 256, sm, stream>>>(
                phi + pko[c], plo + pko[c], W[c], whi + wko[c], wlo + wko[c], B[c],
                vocab[c], dsz[c], kpad[c], S[c],
                pm + (size_t)soff[c] * NTOK, psv + (size_t)soff[c] * NTOK);
        }
    }
    // stage 3: finalize nll
    finalize_kernel<<<NTOK / 256, 256, 0, stream>>>(pm, psv, pf,
                                                    W[0], W[1], W[2], W[3],
                                                    B[0], B[1], B[2], B[3],
                                                    cw, cb, target, out);
}
